// NoisyTopkRouter_cv_9517647528389
// MI455X (gfx1250) — compile-verified, matrix operand order checked by reading, not run on hardware
//
#include <hip/hip_runtime.h>
#include <hip/hip_bf16.h>

#define N_EMBED 2048
#define NUM_EXPERTS 64
#define TOPK 8
#define BATCH 4096
#define SEQ 64

typedef __attribute__((ext_vector_type(16))) __bf16 v16bf;
typedef __attribute__((ext_vector_type(8)))  float  v8f;
typedef __attribute__((ext_vector_type(4)))  float  v4f;

// jax.nn.softplus(x) = max(x,0) + log1p(exp(-|x|))
__device__ __forceinline__ float softplus_f(float x) {
    return fmaxf(x, 0.0f) + log1pf(__expf(-fabsf(x)));
}

// ---------------------------------------------------------------------------
// Prologue: pack W_noise (fp32 [64][2048], row-major) into bf16 B-fragments.
// Fragment layout (symmetric to the ISA 16-bit A layout, B columns = W_noise
// rows are contiguous): packedB[((ct*64 + ks)*32 + lane)*16 + j]
//   e = 16*ct + (lane&15);  o = (lane>=16) ? 8 : 0
//   j<8  -> k = 32*ks + o + j
//   j>=8 -> k = 32*ks + 16 + o + (j-8)
// Total 4*64*32*16 bf16 = 256 KB in d_ws.
// ---------------------------------------------------------------------------
__global__ void pack_wnoise_bf16(const float* __restrict__ Wn,
                                 __bf16* __restrict__ pb) {
    int idx = blockIdx.x * blockDim.x + threadIdx.x;   // 0 .. 131071
    int j    = idx & 15;
    int lane = (idx >> 4) & 31;
    int ks   = (idx >> 9) & 63;
    int ct   = idx >> 15;
    int e = ct * 16 + (lane & 15);
    int o = (lane >= 16) ? 8 : 0;
    int k = ks * 32 + ((j < 8) ? (o + j) : (16 + o + (j - 8)));
    pb[idx] = (__bf16)Wn[e * N_EMBED + k];
}

// ---------------------------------------------------------------------------
// Main kernel: one workgroup per batch row b. 4 waves x 32 lanes.
// Wave w computes the 16x64 slice (token rows 16w..16w+15, all 64 experts)
// of noise_logits via v_wmma_f32_16x16x32_bf16, then softplus + S-reduction.
// Threads also compute the small routing matvec, then noisy combine,
// top-8 and masked softmax.
// ---------------------------------------------------------------------------
__global__ __launch_bounds__(128) void noisy_topk_router_kernel(
    const float* __restrict__ mh,       // [B, S, D]
    const float* __restrict__ mh_avg,   // [B, D]
    const float* __restrict__ eps,      // [B, E]
    const float* __restrict__ Wr,       // [E, D]
    const float* __restrict__ br,       // [E]
    const float* __restrict__ bn,       // [E]
    const __bf16* __restrict__ packedB, // fragment-ordered bf16 W_noise
    float* __restrict__ out_router,     // [B, E]
    float* __restrict__ out_idx)        // [B, K] (float-coded indices)
{
    __shared__ float s_part[4 * NUM_EXPERTS]; // per-wave softplus row sums
    __shared__ float s_logit[NUM_EXPERTS];
    __shared__ float s_noisy[NUM_EXPERTS];

    const int b    = blockIdx.x;
    const int tid  = threadIdx.x;
    const int wave = tid >> 5;
    const int lane = tid & 31;

    // ---------------- GEMM: noise_logits tile for this wave -----------------
    v8f c0 = {}, c1 = {}, c2 = {}, c3 = {};
    const int   mrow  = lane & 15;                 // A row within tile
    const int   o     = (lane >= 16) ? 8 : 0;      // K sub-offset per ISA layout
    const float* Arow = mh + ((size_t)b * SEQ + wave * 16 + mrow) * N_EMBED;

    for (int ks = 0; ks < 64; ++ks) {
        const int k0 = ks * 32;
        // A fragment: 16 fp32 -> bf16 (streamed once: non-temporal loads)
        v4f a0 = __builtin_nontemporal_load((const v4f*)(Arow + k0 + o));
        v4f a1 = __builtin_nontemporal_load((const v4f*)(Arow + k0 + o + 4));
        v4f a2 = __builtin_nontemporal_load((const v4f*)(Arow + k0 + 16 + o));
        v4f a3 = __builtin_nontemporal_load((const v4f*)(Arow + k0 + 16 + o + 4));
        v16bf a;
        a[0]  = (__bf16)a0[0]; a[1]  = (__bf16)a0[1]; a[2]  = (__bf16)a0[2]; a[3]  = (__bf16)a0[3];
        a[4]  = (__bf16)a1[0]; a[5]  = (__bf16)a1[1]; a[6]  = (__bf16)a1[2]; a[7]  = (__bf16)a1[3];
        a[8]  = (__bf16)a2[0]; a[9]  = (__bf16)a2[1]; a[10] = (__bf16)a2[2]; a[11] = (__bf16)a2[3];
        a[12] = (__bf16)a3[0]; a[13] = (__bf16)a3[1]; a[14] = (__bf16)a3[2]; a[15] = (__bf16)a3[3];

        const __bf16* pB = packedB + ((size_t)ks * 32 + lane) * 16; // ct stride = 32768
        v16bf b0 = *(const v16bf*)(pB);
        v16bf b1 = *(const v16bf*)(pB + 32768);
        v16bf b2 = *(const v16bf*)(pB + 65536);
        v16bf b3 = *(const v16bf*)(pB + 98304);

        c0 = __builtin_amdgcn_wmma_f32_16x16x32_bf16(false, a, false, b0, (short)0, c0, false, false);
        c1 = __builtin_amdgcn_wmma_f32_16x16x32_bf16(false, a, false, b1, (short)0, c1, false, false);
        c2 = __builtin_amdgcn_wmma_f32_16x16x32_bf16(false, a, false, b2, (short)0, c2, false, false);
        c3 = __builtin_amdgcn_wmma_f32_16x16x32_bf16(false, a, false, b3, (short)0, c3, false, false);
    }

    // ------------- epilogue: + b_noise, softplus, reduce over S -------------
    const int n = lane & 15;                       // expert within col-tile
    const float bn0 = bn[n], bn1 = bn[16 + n], bn2 = bn[32 + n], bn3 = bn[48 + n];
    float acc0 = 0.f, acc1 = 0.f, acc2 = 0.f, acc3 = 0.f;
#pragma unroll
    for (int r = 0; r < 8; ++r) {
        acc0 += softplus_f(c0[r] + bn0);
        acc1 += softplus_f(c1[r] + bn1);
        acc2 += softplus_f(c2[r] + bn2);
        acc3 += softplus_f(c3[r] + bn3);
    }
    // fold lane halves (rows 0-7 with rows 8-15 of the tile)
    acc0 += __shfl_xor(acc0, 16, 32);
    acc1 += __shfl_xor(acc1, 16, 32);
    acc2 += __shfl_xor(acc2, 16, 32);
    acc3 += __shfl_xor(acc3, 16, 32);
    if (lane < 16) {
        float* sp = s_part + wave * NUM_EXPERTS;
        sp[lane]      = acc0;
        sp[16 + lane] = acc1;
        sp[32 + lane] = acc2;
        sp[48 + lane] = acc3;
    }

    // -------- routing logits: logits[e] = mh_avg[b] . W_route[e] + br[e] ----
    {
        const int e    = tid >> 1;       // 0..63
        const int half = tid & 1;        // split D in two
        const float* av = mh_avg + (size_t)b * N_EMBED + half * (N_EMBED / 2);
        const float* wr = Wr + (size_t)e * N_EMBED + half * (N_EMBED / 2);
        float dot = 0.f;
        for (int i = 0; i < N_EMBED / 8; ++i) {
            v4f x = *(const v4f*)(av + i * 4);
            v4f w = *(const v4f*)(wr + i * 4);
            dot = fmaf(x[0], w[0], dot);
            dot = fmaf(x[1], w[1], dot);
            dot = fmaf(x[2], w[2], dot);
            dot = fmaf(x[3], w[3], dot);
        }
        dot += __shfl_xor(dot, 1, 32);
        if (half == 0) s_logit[e] = dot + br[e];
    }
    __syncthreads();

    // ---------------- noisy logits ----------------
    if (tid < NUM_EXPERTS) {
        float ns = (s_part[tid] + s_part[64 + tid] + s_part[128 + tid] + s_part[192 + tid])
                   * (1.0f / (float)SEQ);
        s_noisy[tid] = s_logit[tid] + eps[(size_t)b * NUM_EXPERTS + tid] * ns;
    }
    __syncthreads();

    // ---------------- top-8, masked softmax, write ----------------
    if (tid == 0) {
        float topv[TOPK];
        int   topi[TOPK];
#pragma unroll
        for (int k = 0; k < TOPK; ++k) {
            float best = -__builtin_inff();
            int   bi   = 0;
            for (int j = 0; j < NUM_EXPERTS; ++j) {
                float v = s_noisy[j];
                if (v > best) { best = v; bi = j; }   // ties -> lowest index (jax top_k)
            }
            topv[k] = best;
            topi[k] = bi;
            s_noisy[bi] = -__builtin_inff();
        }
        const float mx = topv[0];
        float ex[TOPK];
        float sum = 0.f;
#pragma unroll
        for (int k = 0; k < TOPK; ++k) { ex[k] = __expf(topv[k] - mx); sum += ex[k]; }
        const float inv = 1.0f / sum;

        float* orow = out_router + (size_t)b * NUM_EXPERTS;
        for (int j = 0; j < NUM_EXPERTS; ++j) orow[j] = 0.0f;
#pragma unroll
        for (int k = 0; k < TOPK; ++k) orow[topi[k]] = ex[k] * inv;

        float* irow = out_idx + (size_t)b * TOPK;
#pragma unroll
        for (int k = 0; k < TOPK; ++k) irow[k] = (float)topi[k];
    }
}

extern "C" void kernel_launch(void* const* d_in, const int* in_sizes, int n_in,
                              void* d_out, int out_size, void* d_ws, size_t ws_size,
                              hipStream_t stream) {
    const float* mh     = (const float*)d_in[0]; // mh_output     [B,S,D]
    const float* mh_avg = (const float*)d_in[1]; // mh_output_avg [B,D]
    const float* eps    = (const float*)d_in[2]; // noise_eps     [B,E]
    const float* Wr     = (const float*)d_in[3]; // W_route       [E,D]
    const float* br     = (const float*)d_in[4]; // b_route       [E]
    const float* Wn     = (const float*)d_in[5]; // W_noise       [E,D]
    const float* bn     = (const float*)d_in[6]; // b_noise       [E]

    __bf16* packedB = (__bf16*)d_ws;             // 256 KB of scratch
    float*  out_router = (float*)d_out;                               // [B,E]
    float*  out_idx    = out_router + (size_t)BATCH * NUM_EXPERTS;    // [B,K]

    // 1) one-shot fp32 -> bf16 fragment pack of W_noise (stays hot in L2)
    pack_wnoise_bf16<<<(4 * 64 * 32 * 16) / 256, 256, 0, stream>>>(Wn, packedB);

    // 2) fused router kernel: one workgroup (4 wave32) per batch row
    noisy_topk_router_kernel<<<BATCH, 128, 0, stream>>>(
        mh, mh_avg, eps, Wr, br, bn, packedB, out_router, out_idx);
}
